// NeighborNet_21337397526529
// MI455X (gfx1250) — compile-verified, matrix operand order checked by reading, not run on hardware
//
#include <hip/hip_runtime.h>
#include <hip/hip_bf16.h>

typedef _Float16 half8  __attribute__((ext_vector_type(8)));
typedef _Float16 half16 __attribute__((ext_vector_type(16)));
typedef float    float8 __attribute__((ext_vector_type(8)));

#define ELEMS_PER_BLOCK 64
#define THREADS 128
#define N_NBR 7
#define T_NBR 3

__device__ __forceinline__ half16 ld_frag(const _Float16* plo, const _Float16* phi) {
    union { half16 v; half8 h[2]; } u;
    u.h[0] = *(const half8*)plo;
    u.h[1] = *(const half8*)phi;
    return u.v;
}

__device__ __forceinline__ float elu(float v) {
    return v > 0.0f ? v : (__expf(v) - 1.0f);
}

__global__ __launch_bounds__(THREADS)
void neighbor_net_kernel(const float* __restrict__ ego,
                         const float* __restrict__ nbr,
                         const float* __restrict__ tW1, const float* __restrict__ tb1,
                         const float* __restrict__ tW2, const float* __restrict__ tb2,
                         const float* __restrict__ tW3, const float* __restrict__ tb3,
                         const float* __restrict__ oW1, const float* __restrict__ ob1,
                         const float* __restrict__ oW2, const float* __restrict__ ob2,
                         const float* __restrict__ oW3, const float* __restrict__ ob3,
                         float* __restrict__ out, int Btot) {
    // ---- LDS layout (static, ~62 KB) ----
    __shared__ __align__(16) _Float16 s_ego[ELEMS_PER_BLOCK * 16];          // 2 KB
    __shared__ __align__(16) _Float16 s_x[ELEMS_PER_BLOCK * N_NBR * 32];    // 28 KB, rows of [nbr|ego] f16
    __shared__ __align__(16) _Float16 s_w1[2 * 64 * 32];                    // col-major, 8 KB
    __shared__ __align__(16) _Float16 s_w2[2 * 32 * 64];                    // col-major, 8 KB
    __shared__ __align__(16) _Float16 s_w3[2 * 16 * 32];                    // col-major padded, 2 KB
    __shared__ float    s_b[2 * 112];                                      // b1[64] b2[32] b3pad[16] per net
    __shared__ unsigned s_mask[4 * N_NBR];                                 // active bits per (wave-group, slot)
    __shared__ unsigned char s_egoact[ELEMS_PER_BLOCK];
    __shared__ __align__(16) _Float16 s_h1[4 * 16 * 64];                   // per-wave L1 activations
    __shared__ __align__(16) _Float16 s_h2[4 * 16 * 32];                   // per-wave L2 activations

    const int tid = threadIdx.x;
    const int e0  = blockIdx.x * ELEMS_PER_BLOCK;

    // ---- stage ego (f32 -> f16, NaN->0) + per-elem ego-valid flag; zero masks ----
    if (tid < 4 * N_NBR) s_mask[tid] = 0u;
    if (tid < ELEMS_PER_BLOCK) {
        int eg = e0 + tid;
        bool act = true;
        if (eg < Btot) {
            const float* pe = ego + (size_t)eg * 16;
            #pragma unroll
            for (int j = 0; j < 16; ++j) {
                float v = pe[j];
                if (v != v) { act = false; v = 0.0f; }
                s_ego[tid * 16 + j] = (_Float16)v;
            }
        } else {
            act = false;
            #pragma unroll
            for (int j = 0; j < 16; ++j) s_ego[tid * 16 + j] = (_Float16)0.0f;
        }
        s_egoact[tid] = act ? 1 : 0;
    }
    __syncthreads();

    // ---- stage neighbor rows: x = [nbr16 | ego16] f16, NaN->0; build active masks ----
    for (int r = tid; r < ELEMS_PER_BLOCK * N_NBR; r += THREADS) {
        int e = r / N_NBR, s = r - e * N_NBR;
        int eg = e0 + e;
        _Float16* dst = &s_x[r * 32];
        bool act = (s_egoact[e] != 0);
        if (eg < Btot) {
            const float* pn = nbr + (size_t)eg * (N_NBR * 16) + s * 16;
            #pragma unroll
            for (int j = 0; j < 16; ++j) {
                float v = pn[j];
                if (v != v) { act = false; v = 0.0f; }
                dst[j] = (_Float16)v;
            }
        } else {
            act = false;
            #pragma unroll
            for (int j = 0; j < 16; ++j) dst[j] = (_Float16)0.0f;
        }
        #pragma unroll
        for (int j = 0; j < 16; ++j) dst[16 + j] = s_ego[e * 16 + j];
        if (act) atomicOr(&s_mask[(e >> 4) * N_NBR + s], 1u << (e & 15));
    }
    // ---- stage weights (f32 row-major -> f16 column-major in LDS) ----
    for (int idx = tid; idx < 2 * 2048; idx += THREADS) {           // W1: 32x64
        int net = idx >> 11, i = idx & 2047;
        int n = i >> 5, k = i & 31;
        const float* W = net ? oW1 : tW1;
        s_w1[idx] = (_Float16)W[k * 64 + n];
    }
    for (int idx = tid; idx < 2 * 2048; idx += THREADS) {           // W2: 64x32
        int net = idx >> 11, i = idx & 2047;
        int n = i >> 6, k = i & 63;
        const float* W = net ? oW2 : tW2;
        s_w2[idx] = (_Float16)W[k * 32 + n];
    }
    for (int idx = tid; idx < 2 * 512; idx += THREADS) {            // W3: 32x8 padded to 32x16
        int net = idx >> 9, i = idx & 511;
        int n = i >> 5, k = i & 31;
        const float* W = net ? oW3 : tW3;
        s_w3[idx] = (_Float16)((n < 8) ? W[k * 8 + n] : 0.0f);
    }
    for (int idx = tid; idx < 2 * 112; idx += THREADS) {
        int net = idx / 112, j = idx - net * 112;
        float v;
        if (j < 64)       v = (net ? ob1 : tb1)[j];
        else if (j < 96)  v = (net ? ob2 : tb2)[j - 64];
        else if (j < 104) v = (net ? ob3 : tb3)[j - 96];
        else              v = 0.0f;
        s_b[idx] = v;
    }
    __syncthreads();

    // ---- per-wave WMMA pipeline: wave w owns 16 batch elements ----
    const int lane = tid & 31;
    const int w    = tid >> 5;
    const int hi   = (lane >> 4) & 1;
    const int n    = lane & 15;
    const int hoff = hi ? 8 : 0;
    const int ebase = e0 + w * 16;

    #pragma unroll
    for (int phase = 0; phase < 2; ++phase) {
        const int net = phase;
        const int s_begin = phase ? T_NBR : 0;
        const int s_end   = phase ? N_NBR : T_NBR;
        const float* bb = &s_b[net * 112];

        // gather B fragments (column-major weights; lane holds its column's K values)
        half16 B1[4], B2k[2][2], B3;
        #pragma unroll
        for (int f = 0; f < 4; ++f) {
            const _Float16* p = &s_w1[net * 2048 + (f * 16 + n) * 32];
            B1[f] = ld_frag(p + hoff, p + 16 + hoff);
        }
        #pragma unroll
        for (int kb = 0; kb < 2; ++kb)
            #pragma unroll
            for (int j = 0; j < 2; ++j) {
                const _Float16* p = &s_w2[net * 2048 + (j * 16 + n) * 64 + kb * 32];
                B2k[kb][j] = ld_frag(p + hoff, p + 16 + hoff);
            }
        {
            const _Float16* p = &s_w3[net * 512 + n * 32];
            B3 = ld_frag(p + hoff, p + 16 + hoff);
        }

        float m[8];
        #pragma unroll
        for (int i = 0; i < 8; ++i) m[i] = -__builtin_inff();

        for (int s = s_begin; s < s_end; ++s) {
            // ---- layer 1: x[16x32] @ W1 -> h1[16x64] ----
            const _Float16* px = &s_x[((w * 16 + n) * N_NBR + s) * 32];
            half16 a = ld_frag(px + hoff, px + 16 + hoff);
            #pragma unroll
            for (int f = 0; f < 4; ++f) {
                float b = bb[f * 16 + n];
                float8 c = {b, b, b, b, b, b, b, b};
                c = __builtin_amdgcn_wmma_f32_16x16x32_f16(false, a, false, B1[f],
                                                           (short)0, c, false, false);
                #pragma unroll
                for (int i = 0; i < 8; ++i) {
                    float v = elu(c[i]);
                    s_h1[w * 1024 + (i + hoff) * 64 + f * 16 + n] = (_Float16)v;
                }
            }
            __asm__ volatile("s_wait_dscnt 0" ::: "memory");

            // ---- layer 2: h1[16x64] @ W2 -> h2[16x32] (K=64 as 2 chained K=32) ----
            const _Float16* ph = &s_h1[w * 1024 + n * 64];
            half16 a0 = ld_frag(ph + hoff,      ph + 16 + hoff);
            half16 a1 = ld_frag(ph + 32 + hoff, ph + 48 + hoff);
            #pragma unroll
            for (int j = 0; j < 2; ++j) {
                float b = bb[64 + j * 16 + n];
                float8 c = {b, b, b, b, b, b, b, b};
                c = __builtin_amdgcn_wmma_f32_16x16x32_f16(false, a0, false, B2k[0][j],
                                                           (short)0, c, false, false);
                c = __builtin_amdgcn_wmma_f32_16x16x32_f16(false, a1, false, B2k[1][j],
                                                           (short)0, c, false, false);
                #pragma unroll
                for (int i = 0; i < 8; ++i) {
                    float v = elu(c[i]);
                    s_h2[w * 512 + (i + hoff) * 32 + j * 16 + n] = (_Float16)v;
                }
            }
            __asm__ volatile("s_wait_dscnt 0" ::: "memory");

            // ---- layer 3: h2[16x32] @ W3pad -> d[16x16] (cols 0..7 valid) ----
            const _Float16* p2 = &s_h2[w * 512 + n * 32];
            half16 a2 = ld_frag(p2 + hoff, p2 + 16 + hoff);
            float b3v = bb[96 + n];
            float8 c3 = {b3v, b3v, b3v, b3v, b3v, b3v, b3v, b3v};
            c3 = __builtin_amdgcn_wmma_f32_16x16x32_f16(false, a2, false, B3,
                                                        (short)0, c3, false, false);

            // ---- masked running max over neighbor slots ----
            unsigned msk = s_mask[w * N_NBR + s];
            #pragma unroll
            for (int i = 0; i < 8; ++i) {
                float v = ((msk >> (i + hoff)) & 1u) ? c3[i] : -__builtin_inff();
                m[i] = fmaxf(m[i], v);
            }
        }

        if (phase == 0) {
            #pragma unroll
            for (int i = 0; i < 8; ++i)
                if (__builtin_isinf(m[i])) m[i] = -2.0f;
        }

        if (n < 8) {
            #pragma unroll
            for (int i = 0; i < 8; ++i) {
                int row = ebase + i + hoff;
                if (row < Btot) out[(size_t)row * 16 + phase * 8 + n] = m[i];
            }
        }
    }
}

extern "C" void kernel_launch(void* const* d_in, const int* in_sizes, int n_in,
                              void* d_out, int out_size, void* d_ws, size_t ws_size,
                              hipStream_t stream) {
    (void)n_in; (void)d_ws; (void)ws_size; (void)out_size;
    const float* ego = (const float*)d_in[0];
    const float* nbr = (const float*)d_in[1];
    const float* tW1 = (const float*)d_in[2];
    const float* tb1 = (const float*)d_in[3];
    const float* tW2 = (const float*)d_in[4];
    const float* tb2 = (const float*)d_in[5];
    const float* tW3 = (const float*)d_in[6];
    const float* tb3 = (const float*)d_in[7];
    const float* oW1 = (const float*)d_in[8];
    const float* ob1 = (const float*)d_in[9];
    const float* oW2 = (const float*)d_in[10];
    const float* ob2 = (const float*)d_in[11];
    const float* oW3 = (const float*)d_in[12];
    const float* ob3 = (const float*)d_in[13];
    float* out = (float*)d_out;

    int Btot = in_sizes[0] / 16;
    int nblk = (Btot + ELEMS_PER_BLOCK - 1) / ELEMS_PER_BLOCK;
    neighbor_net_kernel<<<nblk, THREADS, 0, stream>>>(
        ego, nbr, tW1, tb1, tW2, tb2, tW3, tb3,
        oW1, ob1, oW2, ob2, oW3, ob3, out, Btot);
}
